// Enhancer_63350767616202
// MI455X (gfx1250) — matrix operand order repaired and checked by reading, not compiled, not run on hardware
//
#include <hip/hip_runtime.h>
#include <hip/hip_bf16.h>
#include <math.h>

// ---------------- problem constants ----------------
#define Bn   8
#define Cc   256
#define Hh   96
#define Ww   192
#define HW   (Hh * Ww)            // 18432
#define Mtot (Bn * HW)            // 147456
#define HID  512
#define F1   (2 * HID)            // 1024
#define DC   64                   // DIM_CONV
#define KPC  (9 * DC)             // 576 (implicit-gemm K for pconv)
#define LN_EPS 1e-5f

#define KC   32                   // K chunk staged in LDS
#define PROW 40                   // padded LDS row (elements): breaks ds bank-conflict stride

typedef __attribute__((ext_vector_type(16))) __bf16 v16bf;
typedef __attribute__((ext_vector_type(8)))  __bf16 v8bf;
typedef __attribute__((ext_vector_type(8)))  float  v8f;
typedef __attribute__((__vector_size__(16))) int    vi4;   // matches async-builtin param type

union FragU { v16bf v; v8bf h[2]; };

__device__ __forceinline__ float gelu_exact(float x) {
    return 0.5f * x * (1.0f + erff(x * 0.70710678118654752f));
}

// ---------------- CDNA5 async global->LDS staging ----------------
#if __has_builtin(__builtin_amdgcn_global_load_async_to_lds_b128)
#define HAS_ASYNC 1
#else
#define HAS_ASYNC 0
#endif

__device__ __forceinline__ void copy16(void* lds_dst, const void* g_src) {
#if HAS_ASYNC
    __builtin_amdgcn_global_load_async_to_lds_b128(
        (__attribute__((address_space(1))) vi4*)(uintptr_t)g_src,
        (__attribute__((address_space(3))) vi4*)(uint32_t)(uintptr_t)lds_dst,
        0, 0);
#else
    *(v8bf*)lds_dst = *(const v8bf*)g_src;
#endif
}

__device__ __forceinline__ void wait_stage() {
#if HAS_ASYNC
#if __has_builtin(__builtin_amdgcn_s_wait_asynccnt)
    __builtin_amdgcn_s_wait_asynccnt(0);
#else
    asm volatile("s_wait_asynccnt 0" ::: "memory");
#endif
#endif
    __syncthreads();
}

// stage a rows x KC bf16 tile (row-major, leading dim ld) into padded LDS
__device__ __forceinline__ void stage_tile(const __bf16* gbase, int ld, int rows,
                                           __bf16* lds, int tid) {
    int nch = rows * (KC / 8);                 // 16B chunks
    for (int c = tid; c < nch; c += 256) {
        int row = c >> 2, o = (c & 3) * 8;
        copy16(lds + row * PROW + o, gbase + (size_t)row * ld + o);
    }
}

// A-matrix fragment (ISA 7.12.2, 16-bit A 16x32): lane = M row; elements 0..7 at
// k = g*8, elements 8..15 at k = 16+g*8 (g = lane/16). Two 16B loads.
__device__ __forceinline__ v16bf lds_fragA(const __bf16* p) {
    FragU f;
    f.h[0] = *(const v8bf*)p;
    f.h[1] = *(const v8bf*)(p + 16);
    return f.v;
}

// B-matrix fragment: lane = N column; elements 0..7 at k = kb + g*8,
// elements 8..15 at k = kb + 16 + g*8 (same order as A). Two 16B loads.
// B is given as B^T rows (N x K row-major), p points at row n, offset kb+g*8.
__device__ __forceinline__ v16bf lds_fragB(const __bf16* p) {
    FragU f;
    f.h[0] = *(const v8bf*)p;
    f.h[1] = *(const v8bf*)(p + 16);
    return f.v;
}

__device__ __forceinline__ v8f wmma_bf16(v16bf a, v16bf b, v8f c) {
    return __builtin_amdgcn_wmma_f32_16x16x32_bf16(false, a, false, b,
                                                   (short)0, c, false, false);
}

// global B fragment load (row-major B^T, two 16B runs per lane at kb+g*8, kb+16+g*8) — pconv weights
__device__ __forceinline__ v16bf load_fragB_g(const __bf16* base, int ld, int kb, int lane) {
    int n = lane & 15, g = lane >> 4;
    const __bf16* p = base + (size_t)n * ld + kb + g * 8;
    FragU f;
    f.h[0] = *(const v8bf*)(p);
    f.h[1] = *(const v8bf*)(p + 16);
    return f.v;
}

// ---------------- weight repack kernels ----------------
__global__ void k_cvt_bf16(const float* __restrict__ src, __bf16* __restrict__ dst, int n) {
    int i = blockIdx.x * 256 + threadIdx.x;
    if (i < n) dst[i] = (__bf16)src[i];
}

// pconv_w (3,3,64,64) HWIO -> wt[co][tap*64+ci] (64 x 576) bf16 row-major
__global__ void k_repack_pconv(const float* __restrict__ pw, __bf16* __restrict__ wt) {
    int i = blockIdx.x * 256 + threadIdx.x;
    if (i < DC * KPC) {
        int co = i / KPC, k = i % KPC;
        wt[i] = (__bf16)pw[(size_t)k * DC + co];
    }
}

// ---------------- LayerNorm (NCHW -> NHWC) ----------------
__global__ void k_ln(const float* __restrict__ X, const float* __restrict__ gam,
                     const float* __restrict__ bet, float* __restrict__ Y64,
                     __bf16* __restrict__ Z) {
    int wave = threadIdx.x >> 5, lane = threadIdx.x & 31;
    int m  = blockIdx.x * 8 + wave;
    int b  = m / HW, hw = m % HW;
    const float* xp = X + (size_t)b * Cc * HW + hw;
    float xv[8], s = 0.f, s2 = 0.f;
#pragma unroll
    for (int i = 0; i < 8; ++i) {
        float v = xp[(size_t)(lane + 32 * i) * HW];
        xv[i] = v; s += v; s2 += v * v;
    }
#pragma unroll
    for (int off = 16; off > 0; off >>= 1) {
        s  += __shfl_xor(s,  off, 32);
        s2 += __shfl_xor(s2, off, 32);
    }
    float mean = s * (1.f / Cc);
    float var  = s2 * (1.f / Cc) - mean * mean;
    float rs   = rsqrtf(var + LN_EPS);
#pragma unroll
    for (int i = 0; i < 8; ++i) {
        int c = lane + 32 * i;
        float yv = (xv[i] - mean) * rs * gam[c] + bet[c];
        if (c < DC) Y64[(size_t)m * DC + c] = yv;
        else        Z[(size_t)m * Cc + c]   = (__bf16)yv;
    }
}

// ---------------- 3x3 conv on first 64 ch: implicit GEMM via WMMA ----------------
__device__ __forceinline__ v8bf conv_chunk(const float* __restrict__ src, bool ok) {
    v8bf out;
    if (ok) {
        const float4* q = (const float4*)src;
        float4 u0 = q[0], u1 = q[1];
        out[0] = (__bf16)u0.x; out[1] = (__bf16)u0.y; out[2] = (__bf16)u0.z; out[3] = (__bf16)u0.w;
        out[4] = (__bf16)u1.x; out[5] = (__bf16)u1.y; out[6] = (__bf16)u1.z; out[7] = (__bf16)u1.w;
    } else {
#pragma unroll
        for (int i = 0; i < 8; ++i) out[i] = (__bf16)0.0f;
    }
    return out;
}

// M=147456, N=64, K=576. 8 waves/block, each wave a 16x64 tile -> 128 rows/block.
__global__ void __launch_bounds__(256) k_pconv(const float* __restrict__ Y64,
                                               const __bf16* __restrict__ Wt,
                                               __bf16* __restrict__ Z) {
    int lane = threadIdx.x & 31, wave = threadIdx.x >> 5;
    int m0 = blockIdx.x * 128 + wave * 16;
    int g  = lane >> 4;
    int mr = m0 + (lane & 15);
    int b  = mr / HW, hw = mr % HW;
    int ph = hw / Ww, pw = hw % Ww;

    v8f acc[4];
#pragma unroll
    for (int j = 0; j < 4; ++j) acc[j] = (v8f)(0.0f);

#pragma unroll
    for (int tap = 0; tap < 9; ++tap) {
        const int ky = tap / 3 - 1, kx = tap % 3 - 1;
        int hh = ph + ky, ww = pw + kx;
        bool ok = (hh >= 0) && (hh < Hh) && (ww >= 0) && (ww < Ww);
        const float* src = Y64 + (((size_t)b * Hh + hh) * Ww + ww) * DC;
#pragma unroll
        for (int cb = 0; cb < DC; cb += 32) {
            FragU a;
            a.h[0] = conv_chunk(src + cb + g * 8, ok);
            a.h[1] = conv_chunk(src + cb + 16 + g * 8, ok);
            int kb = tap * DC + cb;
#pragma unroll
            for (int j = 0; j < 4; ++j) {
                v16bf bf = load_fragB_g(Wt + (size_t)(16 * j) * KPC, KPC, kb, lane);
                acc[j] = wmma_bf16(a.v, bf, acc[j]);
            }
        }
    }
    int nl = lane & 15;
#pragma unroll
    for (int j = 0; j < 4; ++j) {
        int co = 16 * j + nl;
#pragma unroll
        for (int r = 0; r < 8; ++r) {
            int m = m0 + r + g * 8;
            Z[(size_t)m * Cc + co] = (__bf16)acc[j][r];
        }
    }
}

// ---------------- GEMM1: h = gelu(z @ W1^T + b1), bf16 out ----------------
__global__ void __launch_bounds__(256) k_gemm1(const __bf16* __restrict__ Z,
                                               const __bf16* __restrict__ W1,
                                               const float* __restrict__ b1,
                                               __bf16* __restrict__ Hout) {
    __shared__ __bf16 As[2][64 * PROW];
    __shared__ __bf16 Bs[2][256 * PROW];
    int tid = threadIdx.x, lane = tid & 31, wave = tid >> 5;
    int wm = wave >> 2, wn = wave & 3;
    int m0 = blockIdx.x * 64;
    int n0 = blockIdx.y * 256;
    const __bf16* Ab = Z + (size_t)m0 * Cc;
    const __bf16* Bb = W1 + (size_t)n0 * Cc;

    stage_tile(Ab, Cc, 64, As[0], tid);
    stage_tile(Bb, Cc, 256, Bs[0], tid);
    wait_stage();

    v8f acc[8];
#pragma unroll
    for (int i = 0; i < 8; ++i) acc[i] = (v8f)(0.0f);

    int r = lane & 15, g = lane >> 4;
    int buf = 0;
    const int NCH = Cc / KC;                       // 8
    for (int ch = 0; ch < NCH; ++ch) {
        if (ch + 1 < NCH) {
            stage_tile(Ab + (ch + 1) * KC, Cc, 64, As[buf ^ 1], tid);
            stage_tile(Bb + (ch + 1) * KC, Cc, 256, Bs[buf ^ 1], tid);
        }
        v16bf a0 = lds_fragA(&As[buf][(wm * 32 + r) * PROW + g * 8]);
        v16bf a1 = lds_fragA(&As[buf][(wm * 32 + 16 + r) * PROW + g * 8]);
#pragma unroll
        for (int j = 0; j < 4; ++j) {
            v16bf bfr = lds_fragB(&Bs[buf][(wn * 64 + j * 16 + r) * PROW + g * 8]);
            acc[j]     = wmma_bf16(a0, bfr, acc[j]);
            acc[4 + j] = wmma_bf16(a1, bfr, acc[4 + j]);
        }
        wait_stage();
        buf ^= 1;
    }
#pragma unroll
    for (int i = 0; i < 2; ++i)
#pragma unroll
    for (int j = 0; j < 4; ++j) {
        int f = n0 + wn * 64 + j * 16 + r;
        float bias = b1[f];
#pragma unroll
        for (int q = 0; q < 8; ++q) {
            int m = m0 + wm * 32 + i * 16 + q + g * 8;
            Hout[(size_t)m * F1 + f] = (__bf16)gelu_exact(acc[i * 4 + j][q] + bias);
        }
    }
}

// ---------------- depthwise 3x3 + bias + gelu, gate with h2 ----------------
__global__ void k_dw(const __bf16* __restrict__ Hbuf, const float* __restrict__ dww,
                     const float* __restrict__ dwb, __bf16* __restrict__ P) {
    int idx = blockIdx.x * 256 + threadIdx.x;
    int m = idx >> 9, c = idx & 511;
    int b = m / HW, hw = m % HW;
    int h = hw / Ww, w = hw % Ww;
    float acc = 0.f;
#pragma unroll
    for (int ky = -1; ky <= 1; ++ky) {
#pragma unroll
        for (int kx = -1; kx <= 1; ++kx) {
            int hh = h + ky, ww = w + kx;
            if (hh >= 0 && hh < Hh && ww >= 0 && ww < Ww) {
                int m2 = b * HW + hh * Ww + ww;
                acc += (float)Hbuf[(size_t)m2 * F1 + c] * dww[((ky + 1) * 3 + kx + 1) * HID + c];
            }
        }
    }
    acc = gelu_exact(acc + dwb[c]);
    float h2 = (float)Hbuf[(size_t)m * F1 + HID + c];
    P[(size_t)m * HID + c] = (__bf16)(acc * h2);
}

// ---------------- GEMM2: s = xt + p @ W2^T + b2, f32 out ----------------
__global__ void __launch_bounds__(256) k_gemm2(const __bf16* __restrict__ P,
                                               const __bf16* __restrict__ W2,
                                               const float* __restrict__ b2,
                                               const float* __restrict__ X,
                                               float* __restrict__ S) {
    __shared__ __bf16 As[2][64 * PROW];
    __shared__ __bf16 Bs[2][256 * PROW];
    int tid = threadIdx.x, lane = tid & 31, wave = tid >> 5;
    int wm = wave >> 2, wn = wave & 3;
    int m0 = blockIdx.x * 64;
    const __bf16* Ab = P + (size_t)m0 * HID;
    const __bf16* Bb = W2;

    stage_tile(Ab, HID, 64, As[0], tid);
    stage_tile(Bb, HID, 256, Bs[0], tid);
    wait_stage();

    v8f acc[8];
#pragma unroll
    for (int i = 0; i < 8; ++i) acc[i] = (v8f)(0.0f);

    int r = lane & 15, g = lane >> 4;
    int buf = 0;
    const int NCH = HID / KC;                      // 16
    for (int ch = 0; ch < NCH; ++ch) {
        if (ch + 1 < NCH) {
            stage_tile(Ab + (ch + 1) * KC, HID, 64, As[buf ^ 1], tid);
            stage_tile(Bb + (ch + 1) * KC, HID, 256, Bs[buf ^ 1], tid);
        }
        v16bf a0 = lds_fragA(&As[buf][(wm * 32 + r) * PROW + g * 8]);
        v16bf a1 = lds_fragA(&As[buf][(wm * 32 + 16 + r) * PROW + g * 8]);
#pragma unroll
        for (int j = 0; j < 4; ++j) {
            v16bf bfr = lds_fragB(&Bs[buf][(wn * 64 + j * 16 + r) * PROW + g * 8]);
            acc[j]     = wmma_bf16(a0, bfr, acc[j]);
            acc[4 + j] = wmma_bf16(a1, bfr, acc[4 + j]);
        }
        wait_stage();
        buf ^= 1;
    }
#pragma unroll
    for (int i = 0; i < 2; ++i)
#pragma unroll
    for (int j = 0; j < 4; ++j) {
        int c = wn * 64 + j * 16 + r;
        float bias = b2[c];
#pragma unroll
        for (int q = 0; q < 8; ++q) {
            int m = m0 + wm * 32 + i * 16 + q + g * 8;
            int b = m / HW, hw = m % HW;
            float v = acc[i * 4 + j][q] + bias + X[(size_t)b * Cc * HW + (size_t)c * HW + hw];
            S[(size_t)m * Cc + c] = v;
        }
    }
}

// ---------------- spatial mean: gsum[b][c] = sum_hw s ----------------
__global__ void k_red(const float* __restrict__ S, float* __restrict__ gsum) {
    __shared__ float red[256];
    int blk = blockIdx.x;
    int b = blk >> 8, c = blk & 255;
    int tid = threadIdx.x;
    float s = 0.f;
    for (int t = tid; t < HW; t += 256)
        s += S[((size_t)(b * HW + t)) * Cc + c];
    red[tid] = s; __syncthreads();
    for (int off = 128; off > 0; off >>= 1) {
        if (tid < off) red[tid] += red[tid + off];
        __syncthreads();
    }
    if (tid == 0) gsum[b * Cc + c] = red[0];
}

// ---------------- squeeze-excite head ----------------
__global__ void k_se(const float* __restrict__ gsum, const float* __restrict__ fc1,
                     const float* __restrict__ bng, const float* __restrict__ bnb,
                     const float* __restrict__ fc2, float* __restrict__ A) {
    __shared__ float sg[256], st[256], red[256];
    int b = blockIdx.x, c = threadIdx.x;
    sg[c] = gsum[b * Cc + c] * (1.0f / (float)HW);
    __syncthreads();
    float t = 0.f;
    const float* wr = fc1 + (size_t)c * Cc;
    for (int k = 0; k < Cc; ++k) t += sg[k] * wr[k];
    red[c] = t; __syncthreads();
    for (int off = 128; off > 0; off >>= 1) { if (c < off) red[c] += red[c + off]; __syncthreads(); }
    float mean = red[0] * (1.f / Cc);
    __syncthreads();
    float d = t - mean;
    red[c] = d * d; __syncthreads();
    for (int off = 128; off > 0; off >>= 1) { if (c < off) red[c] += red[c + off]; __syncthreads(); }
    float var = red[0] * (1.f / Cc);
    float tn = d * rsqrtf(var + LN_EPS) * bng[c] + bnb[c];
    st[c] = fmaxf(tn, 0.f);
    __syncthreads();
    float u = 0.f;
    const float* w2r = fc2 + (size_t)c * Cc;
    for (int k = 0; k < Cc; ++k) u += st[k] * w2r[k];
    A[b * Cc + c] = 1.0f / (1.0f + expf(-u));
}

// ---------------- out(NCHW) = s(NHWC) * a[b,c] ----------------
__global__ void k_out(const float* __restrict__ S, const float* __restrict__ A,
                      float* __restrict__ Out) {
    __shared__ float tile[16][17];
    int b = blockIdx.z;
    int hw0 = blockIdx.x * 16, c0 = blockIdx.y * 16;
    int tx = threadIdx.x, ty = threadIdx.y;
    tile[ty][tx] = S[((size_t)(b * HW + hw0 + ty)) * Cc + c0 + tx];
    __syncthreads();
    Out[(size_t)b * Cc * HW + (size_t)(c0 + ty) * HW + hw0 + tx] =
        tile[tx][ty] * A[b * Cc + c0 + ty];
}

// ---------------- host launcher ----------------
extern "C" void kernel_launch(void* const* d_in, const int* in_sizes, int n_in,
                              void* d_out, int out_size, void* d_ws, size_t ws_size,
                              hipStream_t stream) {
    const float* x       = (const float*)d_in[0];
    const float* ln2_g   = (const float*)d_in[2];
    const float* ln2_b   = (const float*)d_in[3];
    const float* pconv_w = (const float*)d_in[4];
    const float* lin1_w  = (const float*)d_in[5];
    const float* lin1_b  = (const float*)d_in[6];
    const float* dw_w    = (const float*)d_in[7];
    const float* dw_b    = (const float*)d_in[8];
    const float* lin2_w  = (const float*)d_in[9];
    const float* lin2_b  = (const float*)d_in[10];
    const float* fc1_w   = (const float*)d_in[11];
    const float* bn1_g   = (const float*)d_in[12];
    const float* bn1_b   = (const float*)d_in[13];
    const float* fc2_w   = (const float*)d_in[14];
    float* out = (float*)d_out;

    char* ws = (char*)d_ws;
    size_t off = 0;
    auto carve = [&](size_t bytes) { size_t o = off; off += (bytes + 255) & ~(size_t)255; return o; };
    float*  y64  = (float*)(ws + carve((size_t)Mtot * DC * 4));
    __bf16* z    = (__bf16*)(ws + carve((size_t)Mtot * Cc * 2));
    size_t off_h = carve((size_t)Mtot * F1 * 2);
    __bf16* h    = (__bf16*)(ws + off_h);
    float*  s    = (float*)(ws + off_h);   // alias: h dead once GEMM2 runs
    __bf16* p    = (__bf16*)(ws + carve((size_t)Mtot * HID * 2));
    __bf16* w1b  = (__bf16*)(ws + carve((size_t)F1 * Cc * 2));
    __bf16* w2b  = (__bf16*)(ws + carve((size_t)Cc * HID * 2));
    __bf16* wtp  = (__bf16*)(ws + carve((size_t)DC * KPC * 2));
    float*  gsum = (float*)(ws + carve((size_t)Bn * Cc * 4));
    float*  av   = (float*)(ws + carve((size_t)Bn * Cc * 4));
    (void)ws_size; (void)in_sizes; (void)n_in; (void)out_size;

    k_cvt_bf16<<<(F1 * Cc + 255) / 256, 256, 0, stream>>>(lin1_w, w1b, F1 * Cc);
    k_cvt_bf16<<<(Cc * HID + 255) / 256, 256, 0, stream>>>(lin2_w, w2b, Cc * HID);
    k_repack_pconv<<<(DC * KPC + 255) / 256, 256, 0, stream>>>(pconv_w, wtp);

    k_ln<<<Mtot / 8, 256, 0, stream>>>(x, ln2_g, ln2_b, y64, z);
    k_pconv<<<Mtot / 128, 256, 0, stream>>>(y64, wtp, z);
    k_gemm1<<<dim3(Mtot / 64, F1 / 256), 256, 0, stream>>>(z, w1b, lin1_b, h);
    k_dw<<<(Mtot * HID) / 256, 256, 0, stream>>>(h, dw_w, dw_b, p);
    k_gemm2<<<Mtot / 64, 256, 0, stream>>>(p, w2b, lin2_b, x, s);
    k_red<<<Bn * Cc, 256, 0, stream>>>(s, gsum);
    k_se<<<Bn, 256, 0, stream>>>(gsum, fc1_w, bn1_g, bn1_b, fc2_w, av);
    k_out<<<dim3(HW / 16, Cc / 16, Bn), dim3(16, 16), 0, stream>>>(s, av, out);
}
